// AdaptiveBilateralNetPointwise_29652454212352
// MI455X (gfx1250) — matrix operand order repaired and checked by reading, not compiled, not run on hardware
//
#include <hip/hip_runtime.h>
#include <hip/hip_bf16.h>
#include <math.h>

typedef __attribute__((ext_vector_type(16))) _Float16 v16h;
typedef __attribute__((ext_vector_type(8)))  float    v8f;

#define HW_FULL (1024*1024)

// ---------------------------------------------------------------------------
// 1) Bilinear resize (B,3,1024,1024) -> (B,3,256,256), antialias=False
// ---------------------------------------------------------------------------
__global__ void resize_bilinear_k(const float* __restrict__ in, float* __restrict__ out) {
  int idx = blockIdx.x * blockDim.x + threadIdx.x;
  const int total = 4 * 3 * 256 * 256;
  if (idx >= total) return;
  int x = idx & 255;
  int y = (idx >> 8) & 255;
  int c = idx >> 16;                 // fused (b*3+ch) in [0,12)
  float sx = (x + 0.5f) * 4.0f - 0.5f;
  float sy = (y + 0.5f) * 4.0f - 0.5f;
  float x0f = floorf(sx), y0f = floorf(sy);
  float fx = sx - x0f, fy = sy - y0f;
  int x0 = (int)x0f, y0 = (int)y0f;
  int x0c = min(max(x0, 0), 1023), x1c = min(max(x0 + 1, 0), 1023);
  int y0c = min(max(y0, 0), 1023), y1c = min(max(y0 + 1, 0), 1023);
  const float* p = in + (size_t)c * HW_FULL;
  float v00 = p[y0c * 1024 + x0c], v01 = p[y0c * 1024 + x1c];
  float v10 = p[y1c * 1024 + x0c], v11 = p[y1c * 1024 + x1c];
  out[idx] = (1.f - fy) * ((1.f - fx) * v00 + fx * v01) +
             fy        * ((1.f - fx) * v10 + fx * v11);
}

// ---------------------------------------------------------------------------
// 2) Direct 3x3 stride-2 pad-1 conv + ReLU (tiny layers)
// ---------------------------------------------------------------------------
__global__ void conv3x3_s2_relu_k(const float* __restrict__ in, const float* __restrict__ w,
                                  const float* __restrict__ bias, float* __restrict__ out,
                                  int B, int Cin, int Cout, int Hin, int Win) {
  int Hout = Hin >> 1, Wout = Win >> 1;
  int idx = blockIdx.x * blockDim.x + threadIdx.x;
  int total = B * Cout * Hout * Wout;
  if (idx >= total) return;
  int x  = idx % Wout;
  int y  = (idx / Wout) % Hout;
  int co = (idx / (Wout * Hout)) % Cout;
  int b  = idx / (Wout * Hout * Cout);
  float acc = bias[co];
  for (int ci = 0; ci < Cin; ++ci) {
    const float* ip = in + ((size_t)(b * Cin + ci) * Hin) * Win;
    const float* wp = w + ((size_t)(co * Cin + ci) * 9);
    #pragma unroll
    for (int ky = 0; ky < 3; ++ky) {
      int iy = 2 * y + ky - 1;
      if ((unsigned)iy >= (unsigned)Hin) continue;
      #pragma unroll
      for (int kx = 0; kx < 3; ++kx) {
        int ix = 2 * x + kx - 1;
        if ((unsigned)ix >= (unsigned)Win) continue;
        acc += ip[iy * Win + ix] * wp[ky * 3 + kx];
      }
    }
  }
  out[idx] = fmaxf(acc, 0.0f);
}

// ---------------------------------------------------------------------------
// 3) Pointwise (1x1) conv as WMMA GEMM (compile-time K/N, fully unrolled).
//    out[b,n,p] = sum_k w[n,k]*A[b,k,p] + bias[n] (+val[b]) (opt relu)
//    AFUSE: A element = relu(act + avec[b,k])  (folds the global/local fuse)
//    One wave per 16x16 tile.  npix = 256.
//    Wave32 VGPR striping per CDNA5 ISA (16x16x32 f16 -> f32):
//      A: lanes 0-15 M=lane, K {0..7,16..23}; lanes 16-31 K {8..15,24..31}
//      B: lanes 0-15 N=lane, K {0..7,16..23}; lanes 16-31 K {8..15,24..31}
//      D: vgpr v -> M = v + 8*(lane>=16), N = lane&15
// ---------------------------------------------------------------------------
template <int K, int N, bool RELU, bool ADDVAL, bool AFUSE>
__global__ void __launch_bounds__(32)
pw_gemm_wmma_k(const float* __restrict__ act, const float* __restrict__ w,
               const float* __restrict__ bias, const float* __restrict__ addval,
               const float* __restrict__ avec, float* __restrict__ out) {
  constexpr int NPIX = 256;
  constexpr int NT = N / 16;
  const int lane  = threadIdx.x;
  const int lgrp  = lane >> 4;
  const int lanem = lane & 15;
  int blk   = blockIdx.x;
  int ntile = blk % NT;
  int mtile = (blk / NT) & 15;
  int b     = blk / (NT * 16);

  const int m = mtile * 16 + lanem;            // pixel column this lane holds for A
  const int n = ntile * 16 + lanem;            // out channel this lane holds for B/D
  const float* arow = act + (size_t)b * K * NPIX + m;   // stride NPIX between k
  const float* wrow = w + (size_t)n * K;                // k-contiguous
  const float* vrow = AFUSE ? (avec + b * K) : nullptr;

  v8f c = {};
  #pragma unroll
  for (int k0 = 0; k0 < K; k0 += 32) {
    v16h a, bv;
    // A: two 8-long k-contiguous runs per lane (strided loads, immediate offsets)
    #pragma unroll
    for (int j = 0; j < 8; ++j) {
      #pragma unroll
      for (int t = 0; t < 2; ++t) {
        int ka = (j < 4) ? (lgrp * 8 + 2 * j + t)
                         : (16 + lgrp * 8 + 2 * (j - 4) + t);
        float av = arow[(size_t)(k0 + ka) * NPIX];
        if (AFUSE) av = fmaxf(av + vrow[k0 + ka], 0.0f);
        a[2 * j + t] = (_Float16)av;
      }
    }
    // B: two 8-long k runs (8*lgrp, 16+8*lgrp) -> 4x b128
    {
      const float4* wp = (const float4*)(wrow + k0 + lgrp * 8);
      float4 w0 = wp[0], w1 = wp[1], w2 = wp[4], w3 = wp[5];
      bv[0]  = (_Float16)w0.x; bv[1]  = (_Float16)w0.y; bv[2]  = (_Float16)w0.z; bv[3]  = (_Float16)w0.w;
      bv[4]  = (_Float16)w1.x; bv[5]  = (_Float16)w1.y; bv[6]  = (_Float16)w1.z; bv[7]  = (_Float16)w1.w;
      bv[8]  = (_Float16)w2.x; bv[9]  = (_Float16)w2.y; bv[10] = (_Float16)w2.z; bv[11] = (_Float16)w2.w;
      bv[12] = (_Float16)w3.x; bv[13] = (_Float16)w3.y; bv[14] = (_Float16)w3.z; bv[15] = (_Float16)w3.w;
    }
    c = __builtin_amdgcn_wmma_f32_16x16x32_f16(false, a, false, bv,
                                               (short)0, c, false, false);
  }
  float extra = bias[n] + (ADDVAL ? addval[b] : 0.0f);
  float* orow = out + (size_t)(b * N + n) * NPIX + mtile * 16 + lgrp * 8;
  #pragma unroll
  for (int v = 0; v < 8; ++v) {
    float r = c[v] + extra;
    if (RELU) r = fmaxf(r, 0.0f);
    orow[v] = r;
  }
}

// ---------------------------------------------------------------------------
// 4) Global head: pw 64->4 on splat[::2,::2] + relu, 2x2 mean pool to (B,64),
//    two FC 64->64 with relu.  One block (64 threads) per batch.
// ---------------------------------------------------------------------------
__global__ void global_head_k(const float* __restrict__ splat,
                              const float* __restrict__ cw, const float* __restrict__ cb,
                              const float* __restrict__ fw1, const float* __restrict__ fb1,
                              const float* __restrict__ fw2, const float* __restrict__ fb2,
                              float* __restrict__ cglob) {
  __shared__ float s_sub[64 * 64];
  __shared__ float s_c1[4 * 64];
  __shared__ float s_c[64];
  __shared__ float s_h1[64];
  int b = blockIdx.x, t = threadIdx.x;
  for (int i = t; i < 64 * 64; i += 64) {
    int c = i >> 6, p = i & 63;
    int yy = p >> 3, xx = p & 7;
    s_sub[i] = splat[(size_t)(b * 64 + c) * 256 + (2 * yy) * 16 + 2 * xx];
  }
  __syncthreads();
  for (int o = 0; o < 4; ++o) {
    float acc = cb[o];
    for (int c = 0; c < 64; ++c) acc += cw[o * 64 + c] * s_sub[c * 64 + t];
    s_c1[o * 64 + t] = fmaxf(acc, 0.0f);
  }
  __syncthreads();
  {
    int o = t >> 4, i = (t >> 2) & 3, j = t & 3;
    float s = 0.0f;
    #pragma unroll
    for (int dy = 0; dy < 2; ++dy)
      #pragma unroll
      for (int dx = 0; dx < 2; ++dx)
        s += s_c1[o * 64 + (2 * i + dy) * 8 + (2 * j + dx)];
    s_c[t] = 0.25f * s;
  }
  __syncthreads();
  {
    float acc = fb1[t];
    for (int k = 0; k < 64; ++k) acc += fw1[t * 64 + k] * s_c[k];
    s_h1[t] = fmaxf(acc, 0.0f);
  }
  __syncthreads();
  {
    float acc = fb2[t];
    for (int k = 0; k < 64; ++k) acc += fw2[t * 64 + k] * s_h1[k];
    cglob[b * 64 + t] = fmaxf(acc, 0.0f);
  }
}

// ---------------------------------------------------------------------------
// 5) Repack coeff (B,96,16,16) -> packed (B,8,16,16,16pad), channel-contiguous
//    grid[b,cc,z,y,x] = coeff[b, z*12+cc, y, x]
// ---------------------------------------------------------------------------
__global__ void repack_grid_k(const float* __restrict__ coeff, float* __restrict__ packed) {
  int idx = blockIdx.x * blockDim.x + threadIdx.x;
  const int total = 4 * 8 * 256 * 12;
  if (idx >= total) return;
  int cc  = idx % 12;
  int pix = (idx / 12) & 255;
  int z   = (idx / (12 * 256)) & 7;
  int b   = idx / (12 * 256 * 8);
  packed[(size_t)((b * 8 + z) * 256 + pix) * 16 + cc] =
      coeff[(size_t)(b * 96 + z * 12 + cc) * 256 + pix];
}

// ---------------------------------------------------------------------------
// 6) Fused full-res pass: guide + trilinear slice + 3x4 affine apply.
//    One block = one 256-pixel row segment of one batch.  The <=8z*2y*6x grid
//    cells this segment can touch (6 KB) are staged to LDS with coalesced
//    b128 loads; all 24 per-pixel corner reads come from the DS pipe.
// ---------------------------------------------------------------------------
__global__ void __launch_bounds__(256)
guide_slice_apply_k(const float* __restrict__ img, const float* __restrict__ packed,
                    const float* __restrict__ ccm_w, const float* __restrict__ ccm_b,
                    const float* __restrict__ shifts, const float* __restrict__ slopes,
                    const float* __restrict__ prw, const float* __restrict__ prb,
                    float* __restrict__ out) {
  __shared__ float s_grid[8 * 2 * 6 * 16];   // [z][dy][xcell][16ch] = 6 KB

  int blk = blockIdx.x;                       // b*4096 + y*4 + tx
  int tx  = blk & 3;
  int y   = (blk >> 2) & 1023;
  int b   = blk >> 12;
  int xbase = tx << 8;

  // y cells (uniform per block)
  float cy = (y + 0.5f) * (1.0f / 64.0f) - 0.5f;
  float y0f = floorf(cy);
  float wy  = cy - y0f;
  int y0  = (int)y0f;
  int yi0 = min(max(y0, 0), 15);
  int yi1 = min(max(y0 + 1, 0), 15);

  // x cell window (uniform per block): width <= 6
  int xlo = min(max((int)floorf((xbase + 0.5f) * (1.0f / 64.0f) - 0.5f), 0), 15);

  // cooperative stage: 8z * 2y * 6x cells, 4 float4 each = 384 b128 loads
  for (int i = threadIdx.x; i < 8 * 2 * 6 * 4; i += 256) {
    int c4   = i & 3;
    int cell = i >> 2;
    int xc   = cell % 6;
    int dyy  = (cell / 6) & 1;
    int z    = cell / 12;
    int yi   = dyy ? yi1 : yi0;
    int xi   = min(xlo + xc, 15);
    const float4* src =
        (const float4*)(packed + ((size_t)(((b * 8 + z) * 16 + yi) * 16 + xi) << 4)) + c4;
    ((float4*)s_grid)[(cell << 2) + c4] = *src;
  }
  __syncthreads();

  int x   = xbase + threadIdx.x;
  int pix = (y << 10) + x;

  float r  = img[(size_t)(b * 3 + 0) * HW_FULL + pix];
  float g  = img[(size_t)(b * 3 + 1) * HW_FULL + pix];
  float bl = img[(size_t)(b * 3 + 2) * HW_FULL + pix];

  // guide: ccm -> 16-knot curve -> 3->1 projection -> clip
  float t3[3];
  #pragma unroll
  for (int i = 0; i < 3; ++i) {
    float gi = ccm_w[i * 3 + 0] * r + ccm_w[i * 3 + 1] * g + ccm_w[i * 3 + 2] * bl + ccm_b[i];
    float s = 0.0f;
    #pragma unroll
    for (int k = 0; k < 16; ++k)
      s += slopes[i * 16 + k] * fmaxf(gi - shifts[i * 16 + k], 0.0f);
    t3[i] = s;
  }
  float guide = prw[0] * t3[0] + prw[1] * t3[1] + prw[2] * t3[2] + prb[0];
  guide = fminf(fmaxf(guide, 0.0f), 1.0f);

  // slice coords
  float cx = (x + 0.5f) * (1.0f / 64.0f) - 0.5f;
  float cz = guide * 8.0f - 0.5f;
  float x0f = floorf(cx), z0f = floorf(cz);
  float wx = cx - x0f, wz = cz - z0f;
  int x0 = (int)x0f, z0 = (int)z0f;

  float acc[12];
  #pragma unroll
  for (int c = 0; c < 12; ++c) acc[c] = 0.0f;

  #pragma unroll
  for (int dz = 0; dz < 2; ++dz) {
    int zi = min(max(z0 + dz, 0), 7);
    float wzz = dz ? wz : (1.0f - wz);
    #pragma unroll
    for (int dy = 0; dy < 2; ++dy) {
      float wyy = dy ? wy : (1.0f - wy);
      #pragma unroll
      for (int dx = 0; dx < 2; ++dx) {
        int ci = min(max(x0 + dx, 0), 15) - xlo;   // in [0,6)
        float wxx = dx ? wx : (1.0f - wx);
        float wgt = wzz * wyy * wxx;
        const float4* sp = (const float4*)(s_grid + (((zi * 2 + dy) * 6 + ci) << 4));
        float4 g0 = sp[0], g1 = sp[1], g2 = sp[2];
        acc[0] += wgt * g0.x; acc[1]  += wgt * g0.y; acc[2]  += wgt * g0.z; acc[3]  += wgt * g0.w;
        acc[4] += wgt * g1.x; acc[5]  += wgt * g1.y; acc[6]  += wgt * g1.z; acc[7]  += wgt * g1.w;
        acc[8] += wgt * g2.x; acc[9]  += wgt * g2.y; acc[10] += wgt * g2.z; acc[11] += wgt * g2.w;
      }
    }
  }

  out[(size_t)(b * 3 + 0) * HW_FULL + pix] = acc[0] * r + acc[1] * g + acc[2]  * bl + acc[3];
  out[(size_t)(b * 3 + 1) * HW_FULL + pix] = acc[4] * r + acc[5] * g + acc[6]  * bl + acc[7];
  out[(size_t)(b * 3 + 2) * HW_FULL + pix] = acc[8] * r + acc[9] * g + acc[10] * bl + acc[11];
}

// ---------------------------------------------------------------------------
extern "C" void kernel_launch(void* const* d_in, const int* in_sizes, int n_in,
                              void* d_out, int out_size, void* d_ws, size_t ws_size,
                              hipStream_t stream) {
  (void)in_sizes; (void)n_in; (void)out_size; (void)ws_size;
  const float* image = (const float*)d_in[0];
  const float* val   = (const float*)d_in[1];
  const float* sw0 = (const float*)d_in[2];  const float* sb0 = (const float*)d_in[3];
  const float* sw1 = (const float*)d_in[4];  const float* sb1 = (const float*)d_in[5];
  const float* sw2 = (const float*)d_in[6];  const float* sb2 = (const float*)d_in[7];
  const float* sw3 = (const float*)d_in[8];  const float* sb3 = (const float*)d_in[9];
  const float* spw = (const float*)d_in[10]; const float* spb = (const float*)d_in[11];
  const float* lw1 = (const float*)d_in[12]; const float* lb1 = (const float*)d_in[13];
  const float* lw2 = (const float*)d_in[14]; const float* lb2 = (const float*)d_in[15];
  const float* lw3 = (const float*)d_in[16]; const float* lb3 = (const float*)d_in[17];
  const float* cw  = (const float*)d_in[18]; const float* cb  = (const float*)d_in[19];
  const float* fw1 = (const float*)d_in[20]; const float* fb1 = (const float*)d_in[21];
  const float* fw2 = (const float*)d_in[22]; const float* fb2 = (const float*)d_in[23];
  const float* gw  = (const float*)d_in[24]; const float* gb  = (const float*)d_in[25];
  const float* ccm_w = (const float*)d_in[26]; const float* ccm_b = (const float*)d_in[27];
  const float* shifts = (const float*)d_in[28]; const float* slopes = (const float*)d_in[29];
  const float* prw = (const float*)d_in[30]; const float* prb = (const float*)d_in[31];
  float* outp = (float*)d_out;

  float* ws = (float*)d_ws;
  size_t off = 0;
  float* low    = ws + off; off += (size_t)4 * 3 * 256 * 256;
  float* c1buf  = ws + off; off += (size_t)4 * 8 * 128 * 128;
  float* c2buf  = ws + off; off += (size_t)4 * 16 * 64 * 64;
  float* c3buf  = ws + off; off += (size_t)4 * 32 * 32 * 32;
  float* c4buf  = ws + off; off += (size_t)4 * 64 * 256;
  float* splat  = ws + off; off += (size_t)4 * 64 * 256;
  float* loc1   = ws + off; off += (size_t)4 * 128 * 256;
  float* loc2   = ws + off; off += (size_t)4 * 128 * 256;
  float* loc3   = ws + off; off += (size_t)4 * 64 * 256;
  float* cglob  = ws + off; off += (size_t)256;
  float* coeff  = ws + off; off += (size_t)4 * 96 * 256;
  float* packed = ws + off; off += (size_t)4 * 8 * 256 * 16;

  // 1) downsample
  {
    int total = 4 * 3 * 256 * 256;
    resize_bilinear_k<<<(total + 255) / 256, 256, 0, stream>>>(image, low);
  }
  // 2) strided convs
  {
    int t1 = 4 * 8 * 128 * 128;
    conv3x3_s2_relu_k<<<(t1 + 255) / 256, 256, 0, stream>>>(low, sw0, sb0, c1buf, 4, 3, 8, 256, 256);
    int t2 = 4 * 16 * 64 * 64;
    conv3x3_s2_relu_k<<<(t2 + 255) / 256, 256, 0, stream>>>(c1buf, sw1, sb1, c2buf, 4, 8, 16, 128, 128);
    int t3 = 4 * 32 * 32 * 32;
    conv3x3_s2_relu_k<<<(t3 + 255) / 256, 256, 0, stream>>>(c2buf, sw2, sb2, c3buf, 4, 16, 32, 64, 64);
    int t4 = 4 * 64 * 16 * 16;
    conv3x3_s2_relu_k<<<(t4 + 255) / 256, 256, 0, stream>>>(c3buf, sw3, sb3, c4buf, 4, 32, 64, 32, 32);
  }
  // 3) WMMA pointwise stack (one wave per 16x16 tile, K fully unrolled)
  pw_gemm_wmma_k<64, 64, false, true, false>
      <<<4 * 16 * 4, 32, 0, stream>>>(c4buf, spw, spb, val, nullptr, splat);
  pw_gemm_wmma_k<64, 128, true, false, false>
      <<<4 * 16 * 8, 32, 0, stream>>>(splat, lw1, lb1, nullptr, nullptr, loc1);
  pw_gemm_wmma_k<128, 128, true, false, false>
      <<<4 * 16 * 8, 32, 0, stream>>>(loc1, lw2, lb2, nullptr, nullptr, loc2);
  pw_gemm_wmma_k<128, 64, true, false, false>
      <<<4 * 16 * 4, 32, 0, stream>>>(loc2, lw3, lb3, nullptr, nullptr, loc3);
  // 4) global head
  global_head_k<<<4, 64, 0, stream>>>(splat, cw, cb, fw1, fb1, fw2, fb2, cglob);
  // 5) grid coeffs with fused relu(cglob + loc3) on the A operand
  pw_gemm_wmma_k<64, 96, false, false, true>
      <<<4 * 16 * 6, 32, 0, stream>>>(loc3, gw, gb, nullptr, cglob, coeff);
  // 6) repack grid channel-contiguous (padded to 16 floats)
  {
    int total = 4 * 8 * 256 * 12;
    repack_grid_k<<<(total + 255) / 256, 256, 0, stream>>>(coeff, packed);
  }
  // 7) fused full-res guide + slice + apply with LDS-staged grid window
  guide_slice_apply_k<<<4 * 1024 * 4, 256, 0, stream>>>(
      image, packed, ccm_w, ccm_b, shifts, slopes, prw, prb, outp);
}
